// DoubleBinaryLinear_1958505087324
// MI455X (gfx1250) — matrix operand order repaired and checked by reading, not compiled, not run on hardware
//
#include <hip/hip_runtime.h>
#include <hip/hip_bf16.h>

// MI455X / gfx1250, wave32. Two fused-epilogue f16 WMMA GEMMs:
//   stage1: H[t,m] = (sum_k (x[t,k]*Db[k]) * B[m,k]) * scale1 * mid[m]   (f16 -> d_ws)
//   stage2: out[t,n] = (sum_m H[t,m] * A[n,m]) * scale3 * Da[n] + bias[n] (f32 -> d_out)
// Block tile 128x256, 8 wave32s (2x4), wave tile 64x64 = 4x4 WMMA 16x16x32 tiles.

typedef __attribute__((ext_vector_type(16))) _Float16 v16h;
typedef __attribute__((ext_vector_type(8)))  _Float16 v8h;
typedef __attribute__((ext_vector_type(8)))  float    v8f;

#define DIM  4096      // T = IN = MID = OUT
#define BM   128
#define BN   256
#define BK   32
#define LDSS 40        // 32 + 8 halves pad; row stride 80B keeps 16B-aligned chunks

union HFrag { v16h v; v8h h[2]; };

#if __has_builtin(__builtin_amdgcn_global_load_async_to_lds_b128)
#define HAVE_ASYNC_LDS 1
// builtin signature: (int4 addrspace(1)*, int4 addrspace(3)*, Imm offset, Imm cpol)
typedef int async_v4i __attribute__((vector_size(16)));
typedef __attribute__((address_space(1))) async_v4i* g_v4i_p;
typedef __attribute__((address_space(3))) async_v4i* l_v4i_p;
#endif

__device__ __forceinline__ void wait_asynccnt0() {
#if __has_builtin(__builtin_amdgcn_s_wait_asynccnt)
  __builtin_amdgcn_s_wait_asynccnt(0);
#else
  asm volatile("s_wait_asynccnt 0" ::: "memory");
#endif
}

template<bool ACT_F16, bool FINAL>
__launch_bounds__(256, 1)
__global__ void bitlinear_gemm(const void* __restrict__ act,     // [DIM,DIM] f32 (stage1) or f16 (stage2)
                               const float* __restrict__ kscale, // per-K scale (Db) or nullptr
                               const float* __restrict__ wmat,   // [DIM,DIM] f32, entries +-1
                               const float* __restrict__ nscale, // per-N scale (mid or Da)
                               const float* __restrict__ sscale, // scalar scale (scale1 / scale3)
                               const float* __restrict__ bias,   // [DIM] (stage2) or nullptr
                               void* __restrict__ outp)          // f16 (stage1) or f32 (stage2)
{
  __shared__ _Float16 As[BM][LDSS];
  __shared__ _Float16 Bs[BN][LDSS];

  const int tid  = threadIdx.x;
  const int row0 = blockIdx.y * BM;
  const int col0 = blockIdx.x * BN;

  // 8 waves: 2 along M x 4 along N; each wave -> 64x64 = 4x4 WMMA tiles
  const int wave = tid >> 5;
  const int lane = tid & 31;
  const int wm   = (wave >> 2) * 64;
  const int wn   = (wave & 3)  * 64;
  const int l    = lane & 15;
  const int hi   = lane >> 4;

  // cooperative tile loads: each thread owns 16 consecutive K of one row
  const int lr = tid >> 1;          // 0..127
  const int lk = (tid & 1) * 16;    // 0 or 16

  const v8f vzero = {0.f,0.f,0.f,0.f,0.f,0.f,0.f,0.f};
  v8f acc[4][4];
  #pragma unroll
  for (int mt = 0; mt < 4; ++mt)
    #pragma unroll
    for (int nt = 0; nt < 4; ++nt)
      acc[mt][nt] = vzero;

  for (int k0 = 0; k0 < DIM; k0 += BK) {
    __syncthreads();

    // ---- activation tile (128 x 32) -> LDS ----
    if constexpr (ACT_F16) {
      const _Float16* srcA = (const _Float16*)act + (size_t)(row0 + lr) * DIM + k0 + lk;
#ifdef HAVE_ASYNC_LDS
      // pure f16 copy: route through the async-DMA-to-LDS path (ASYNCcnt)
      __builtin_amdgcn_global_load_async_to_lds_b128(
          (g_v4i_p)srcA, (l_v4i_p)&As[lr][lk], 0, 0);
      __builtin_amdgcn_global_load_async_to_lds_b128(
          (g_v4i_p)srcA, (l_v4i_p)&As[lr][lk], 16, 0);
#else
      *(uint4*)&As[lr][lk]     = ((const uint4*)srcA)[0];
      *(uint4*)&As[lr][lk + 8] = ((const uint4*)srcA)[1];
#endif
      if (k0 + BK < DIM) __builtin_prefetch(srcA + BK, 0, 0);
    } else {
      const float* basep = (const float*)act + (size_t)(row0 + lr) * DIM + k0 + lk;
      const float4* srcA = (const float4*)basep;
      const float4* scl  = (const float4*)(kscale + k0 + lk);
      #pragma unroll
      for (int i = 0; i < 4; ++i) {
        float4 v = srcA[i], s = scl[i];
        As[lr][lk + 4*i + 0] = (_Float16)(v.x * s.x);
        As[lr][lk + 4*i + 1] = (_Float16)(v.y * s.y);
        As[lr][lk + 4*i + 2] = (_Float16)(v.z * s.z);
        As[lr][lk + 4*i + 3] = (_Float16)(v.w * s.w);
      }
      if (k0 + BK < DIM) __builtin_prefetch(basep + BK, 0, 0);
    }

    // ---- weight tile (256 x 32): rows lr and lr+128; f32 +-1 -> f16 ----
    #pragma unroll
    for (int half = 0; half < 2; ++half) {
      const int n = lr + half * 128;
      const float* baseb = wmat + (size_t)(col0 + n) * DIM + k0 + lk;
      const float4* srcB = (const float4*)baseb;
      #pragma unroll
      for (int i = 0; i < 4; ++i) {
        float4 v = srcB[i];
        Bs[n][lk + 4*i + 0] = (_Float16)v.x;
        Bs[n][lk + 4*i + 1] = (_Float16)v.y;
        Bs[n][lk + 4*i + 2] = (_Float16)v.z;
        Bs[n][lk + 4*i + 3] = (_Float16)v.w;
      }
      if (k0 + BK < DIM) __builtin_prefetch(baseb + BK, 0, 0);
    }

#ifdef HAVE_ASYNC_LDS
    if constexpr (ACT_F16) wait_asynccnt0();
#endif
    __syncthreads();

    // ---- load ALL fragments first (distinct VGPRs -> overlapped DS waits) ----
    HFrag a[4], b[4];
    #pragma unroll
    for (int nt = 0; nt < 4; ++nt) {
      const int n = wn + nt * 16 + l;            // column, K = {hi*8..}, {16+hi*8..}
      b[nt].h[0] = *(const v8h*)&Bs[n][hi * 8];
      b[nt].h[1] = *(const v8h*)&Bs[n][16 + hi * 8];
    }
    #pragma unroll
    for (int mt = 0; mt < 4; ++mt) {
      const int m = wm + mt * 16 + l;            // row, K = {hi*8..}, {16+hi*8..}
      a[mt].h[0] = *(const v8h*)&As[m][hi * 8];
      a[mt].h[1] = *(const v8h*)&As[m][16 + hi * 8];
    }

    // ---- 16 WMMAs per K-step per wave ----
    #pragma unroll
    for (int mt = 0; mt < 4; ++mt)
      #pragma unroll
      for (int nt = 0; nt < 4; ++nt)
        acc[mt][nt] = __builtin_amdgcn_wmma_f32_16x16x32_f16(
            false, a[mt].v, false, b[nt].v, (short)0, acc[mt][nt], false, false);
  }

  // ---- fused epilogue ----
  const float sc = sscale[0];
  #pragma unroll
  for (int mt = 0; mt < 4; ++mt) {
    #pragma unroll
    for (int nt = 0; nt < 4; ++nt) {
      const int gr0 = row0 + wm + mt * 16 + hi * 8;   // C/D layout: M = hi*8 + r
      const int gc  = col0 + wn + nt * 16 + l;        //             N = l
      const float cs = sc * nscale[gc];
      if constexpr (FINAL) {
        float* out = (float*)outp;
        const float bz = bias[gc];
        #pragma unroll
        for (int r = 0; r < 8; ++r)
          out[(size_t)(gr0 + r) * DIM + gc] = acc[mt][nt][r] * cs + bz;
      } else {
        _Float16* out = (_Float16*)outp;
        #pragma unroll
        for (int r = 0; r < 8; ++r)
          out[(size_t)(gr0 + r) * DIM + gc] = (_Float16)(acc[mt][nt][r] * cs);
      }
    }
  }
}

extern "C" void kernel_launch(void* const* d_in, const int* in_sizes, int n_in,
                              void* d_out, int out_size, void* d_ws, size_t ws_size,
                              hipStream_t stream) {
  const float* x      = (const float*)d_in[0];
  const float* Db     = (const float*)d_in[1];
  const float* B      = (const float*)d_in[2];
  const float* scale1 = (const float*)d_in[3];
  const float* mid    = (const float*)d_in[4];
  const float* A      = (const float*)d_in[5];
  const float* scale3 = (const float*)d_in[6];
  const float* Da     = (const float*)d_in[7];
  const float* bias   = (const float*)d_in[8];

  _Float16* H = (_Float16*)d_ws;   // 4096*4096 f16 = 32 MiB intermediate

  dim3 grid(DIM / BN, DIM / BM);   // 16 x 32 blocks
  dim3 block(256);                  // 8 wave32s

  // stage 1: H = (x .* Db) @ B^T * scale1 .* mid   (f16)
  bitlinear_gemm<false, false><<<grid, block, 0, stream>>>(
      x, Db, B, mid, scale1, nullptr, (void*)H);

  // stage 2: out = H @ A^T * scale3 .* Da + bias   (f32)
  bitlinear_gemm<true, true><<<grid, block, 0, stream>>>(
      (const void*)H, nullptr, A, Da, scale3, bias, d_out);
}
